// GAMBlock_63728724738161
// MI455X (gfx1250) — matrix operand order repaired and checked by reading, not compiled, not run on hardware
//
#include <hip/hip_runtime.h>

// ---------- types ----------
typedef __attribute__((ext_vector_type(16))) __bf16 v16bf;
typedef __attribute__((ext_vector_type(8)))  __bf16 bf16x8;
typedef __attribute__((ext_vector_type(4)))  __bf16 bf16x4;
typedef __attribute__((ext_vector_type(8)))  float  v8f;

#define CC   512            // channels C
#define NTOK 65536          // B*T
#define TT   8192           // T (per batch)

// ---------- WMMA helpers ----------
__device__ __forceinline__ v8f wmma_bf16f32(v16bf a, v16bf b, v8f c) {
  // D = A(16x32 bf16) * B(32x16 bf16) + C(16x16 f32)
  return __builtin_amdgcn_wmma_f32_16x16x32_bf16(false, a, false, b, (short)0, c,
                                                 false, false);
}

// A fragment: 16x32 bf16, row-major source with row pitch ldk (elements).
// ISA layout: lane<16 -> row M=lane, K = {k0..k0+7, k0+16..k0+23}
//             lane>=16 -> row M=lane-16, K = {k0+8..k0+15, k0+24..k0+31}
__device__ __forceinline__ v16bf load_a_frag(const __bf16* A, int ldk, int m0,
                                             int k0, int lane) {
  const int half = lane >> 4, l = lane & 15;
  const __bf16* p = A + (size_t)(m0 + l) * ldk + k0 + half * 8;
  bf16x8 lo = *(const bf16x8*)(p);
  bf16x8 hi = *(const bf16x8*)(p + 16);
  v16bf r;
#pragma unroll
  for (int i = 0; i < 8; ++i) { r[i] = lo[i]; r[i + 8] = hi[i]; }
  return r;
}

// B fragment: 32x16 bf16. Wt is stored transposed: Wt[n][k], row pitch ldk.
// ISA layout: lane<16 -> column N=n0+lane, K = {k0..k0+7, k0+16..k0+23}
//             lane>=16 -> column N=n0+lane-16, K = {k0+8..k0+15, k0+24..k0+31}
__device__ __forceinline__ v16bf load_b_frag(const __bf16* Wt, int ldk, int n0,
                                             int k0, int lane) {
  const int half = lane >> 4, l = lane & 15;
  const __bf16* p = Wt + (size_t)(n0 + l) * ldk + k0 + half * 8;
  bf16x8 lo = *(const bf16x8*)(p);
  bf16x8 hi = *(const bf16x8*)(p + 16);
  v16bf r;
#pragma unroll
  for (int i = 0; i < 8; ++i) { r[i] = lo[i]; r[i + 8] = hi[i]; }
  return r;
}

__device__ __forceinline__ float wave_sum(float v) {
#pragma unroll
  for (int m = 16; m >= 1; m >>= 1) v += __shfl_xor(v, m, 32);
  return v;
}

// CDNA5 async global->LDS copy (ASYNCcnt-tracked; no VGPR data round-trip).
// VDST VGPR = LDS byte address, VADDR = 64-bit global address (GV mode).
__device__ __forceinline__ void async_g2l_b128(unsigned lds_addr,
                                               const void* gaddr) {
  asm volatile("global_load_async_to_lds_b128 %0, %1, off"
               :: "v"(lds_addr), "v"(gaddr)
               : "memory");
}
__device__ __forceinline__ void wait_async0() {
  asm volatile("s_wait_asynccnt 0x0" ::: "memory");
}

// ---------- weight prep ----------
__global__ void cvt_bf16_kernel(const float* __restrict__ s,
                                __bf16* __restrict__ d, int n) {
  int i = blockIdx.x * blockDim.x + threadIdx.x;
  if (i < n) d[i] = (__bf16)s[i];
}

// d[c*R + r] = s[r*Cc + c]  (store transposed, bf16)
__global__ void transpose_bf16_kernel(const float* __restrict__ s,
                                      __bf16* __restrict__ d, int R, int Cc) {
  int i = blockIdx.x * blockDim.x + threadIdx.x;
  if (i < R * Cc) {
    int r = i / Cc, c = i % Cc;
    d[(size_t)c * R + r] = (__bf16)s[i];
  }
}

// ---------- LayerNorm1: one wave per row, bf16 out ----------
__global__ void __launch_bounds__(256)
ln1_kernel(const float* __restrict__ x, const float* __restrict__ g,
           const float* __restrict__ b, __bf16* __restrict__ out) {
  const int row  = (blockIdx.x * blockDim.x + threadIdx.x) >> 5;
  const int lane = threadIdx.x & 31;
  const float4* rp = (const float4*)(x + (size_t)row * CC);
  float4 v[4];
  float s = 0.f;
#pragma unroll
  for (int i = 0; i < 4; ++i) {
    v[i] = rp[i * 32 + lane];
    s += v[i].x + v[i].y + v[i].z + v[i].w;
  }
  s = wave_sum(s);
  const float mean = s * (1.f / CC);
  float q = 0.f;
#pragma unroll
  for (int i = 0; i < 4; ++i) {
    float dx = v[i].x - mean, dy = v[i].y - mean, dz = v[i].z - mean,
          dw = v[i].w - mean;
    q += dx * dx + dy * dy + dz * dz + dw * dw;
  }
  q = wave_sum(q);
  const float rstd = rsqrtf(q * (1.f / CC) + 1e-5f);
  __bf16* orow = out + (size_t)row * CC;
#pragma unroll
  for (int i = 0; i < 4; ++i) {
    const int c = (i * 32 + lane) * 4;
    bf16x4 o;
    o[0] = (__bf16)((v[i].x - mean) * rstd * g[c + 0] + b[c + 0]);
    o[1] = (__bf16)((v[i].y - mean) * rstd * g[c + 1] + b[c + 1]);
    o[2] = (__bf16)((v[i].z - mean) * rstd * g[c + 2] + b[c + 2]);
    o[3] = (__bf16)((v[i].w - mean) * rstd * g[c + 3] + b[c + 3]);
    *(bf16x4*)(orow + c) = o;
  }
}

// ---------- fused memory attention: scores -> softmax -> ctx, all in LDS ----
// One block handles 16 token rows. 8 waves each own a 16x64 slice.
// The 16x512 bf16 A tile is async-staged to LDS once (kills the 8x redundant
// per-wave global A loads), then both WMMA passes read A from LDS.
__global__ void __launch_bounds__(256)
memattn_kernel(const __bf16* __restrict__ xnorm, const __bf16* __restrict__ mb,
               const __bf16* __restrict__ mbT, __bf16* __restrict__ gctx) {
  __shared__ alignas(16) __bf16 sA[16 * 512];  // staged x_norm tile (16 KB)
  __shared__ alignas(16) float  sS[16 * 512];  // scores / exp, f32 (32 KB)
  __shared__ alignas(16) __bf16 sP[16 * 512];  // softmax probs, bf16 (16 KB)
  const int tid = threadIdx.x, wave = tid >> 5, lane = tid & 31;
  const int r0 = blockIdx.x * 16;
  const int l = lane & 15, mh = (lane >> 4) * 8;

  // phase 0: async-stage A tile (16 KB -> 64 B per thread, 4x b128 each)
  {
    const char* gsrc = (const char*)(xnorm + (size_t)r0 * CC) + tid * 64;
    const unsigned ldst = (unsigned)(uintptr_t)((char*)sA + tid * 64);
#pragma unroll
    for (int i = 0; i < 4; ++i)
      async_g2l_b128(ldst + i * 16, gsrc + i * 16);
    wait_async0();
  }
  __syncthreads();

  // phase 1: S = A(16xC) * mem^T  -> WMMA, B-frag = mb rows (K=c contiguous)
  {
    const int n0 = wave * 64;
    v8f acc[4] = {};
    for (int k0 = 0; k0 < CC; k0 += 32) {
      v16bf a = load_a_frag(sA, CC, 0, k0, lane);
#pragma unroll
      for (int t = 0; t < 4; ++t) {
        v16bf bf = load_b_frag(mb, CC, n0 + t * 16, k0, lane);
        acc[t] = wmma_bf16f32(a, bf, acc[t]);
      }
    }
#pragma unroll
    for (int t = 0; t < 4; ++t)
#pragma unroll
      for (int v = 0; v < 8; ++v)
        sS[(mh + v) * 512 + n0 + t * 16 + l] = acc[t][v];
  }
  __syncthreads();

  // phase 2: row softmax over M=512 (16 threads per row)
  {
    const int r = tid >> 4, sub = tid & 15;
    float mx = -1e30f;
    for (int j = 0; j < 32; ++j) mx = fmaxf(mx, sS[r * 512 + sub + j * 16]);
#pragma unroll
    for (int m = 8; m >= 1; m >>= 1) mx = fmaxf(mx, __shfl_xor(mx, m, 16));
    float s = 0.f;
    for (int j = 0; j < 32; ++j) {
      float e = __expf(sS[r * 512 + sub + j * 16] - mx);
      sS[r * 512 + sub + j * 16] = e;
      s += e;
    }
#pragma unroll
    for (int m = 8; m >= 1; m >>= 1) s += __shfl_xor(s, m, 16);
    const float inv = 1.f / s;
    for (int j = 0; j < 32; ++j)
      sP[r * 512 + sub + j * 16] = (__bf16)(sS[r * 512 + sub + j * 16] * inv);
  }
  __syncthreads();

  // phase 3: ctx = P(16xM) * mem(MxC) -> WMMA, B-frag from mbT (K=m contiguous)
  {
    const int n0 = wave * 64;
    v8f acc[4] = {};
    for (int k0 = 0; k0 < 512; k0 += 32) {
      v16bf a = load_a_frag(sP, 512, 0, k0, lane);
#pragma unroll
      for (int t = 0; t < 4; ++t) {
        v16bf bf = load_b_frag(mbT, 512, n0 + t * 16, k0, lane);
        acc[t] = wmma_bf16f32(a, bf, acc[t]);
      }
    }
#pragma unroll
    for (int t = 0; t < 4; ++t)
#pragma unroll
      for (int v = 0; v < 8; ++v)
        gctx[(size_t)(r0 + mh + v) * CC + n0 + t * 16 + l] = (__bf16)acc[t][v];
  }
}

// ---------- generic bf16 WMMA GEMM with fused epilogue ----------
// Y[r,n] = epilogue( sum_k A[r,k]*Wt[n,k] + bias[n] )
// MODE 0: sigmoid -> bf16     MODE 1: exact gelu -> bf16
// MODE 2: + res[r,n] -> f32
template <int MODE>
__global__ void __launch_bounds__(256)
gemm_bf16_kernel(const __bf16* __restrict__ A, const __bf16* __restrict__ Wt,
                 const float* __restrict__ bias, const float* __restrict__ res,
                 __bf16* __restrict__ outb, float* __restrict__ outf, int K,
                 int Ncols) {
  const int wave = threadIdx.x >> 5, lane = threadIdx.x & 31;
  const int m0 = blockIdx.x * 128 + wave * 16;
  const int n0 = blockIdx.y * 64;
  v8f acc[4] = {};
  for (int k0 = 0; k0 < K; k0 += 32) {
    v16bf a = load_a_frag(A, K, m0, k0, lane);
#pragma unroll
    for (int t = 0; t < 4; ++t) {
      v16bf b = load_b_frag(Wt, K, n0 + t * 16, k0, lane);
      acc[t] = wmma_bf16f32(a, b, acc[t]);
    }
  }
  const int l = lane & 15, mh = (lane >> 4) * 8;
#pragma unroll
  for (int t = 0; t < 4; ++t) {
    const int n = n0 + t * 16 + l;
    const float bn = bias[n];
#pragma unroll
    for (int v = 0; v < 8; ++v) {
      const size_t idx = (size_t)(m0 + mh + v) * Ncols + n;
      float val = acc[t][v] + bn;
      if (MODE == 0) {
        outb[idx] = (__bf16)(1.f / (1.f + __expf(-val)));
      } else if (MODE == 1) {
        outb[idx] = (__bf16)(0.5f * val * (1.f + erff(val * 0.70710678118654752f)));
      } else {
        outf[idx] = val + res[idx];
      }
    }
  }
}

// ---------- fused: causal depthwise conv + gating + residual + LN2 ----------
__global__ void __launch_bounds__(256)
combine_ln2_kernel(const float* __restrict__ x, const __bf16* __restrict__ xnorm,
                   const __bf16* __restrict__ gctx, const __bf16* __restrict__ sig,
                   const float* __restrict__ conv_w, const float* __restrict__ conv_b,
                   const float* __restrict__ g2, const float* __restrict__ b2,
                   float* __restrict__ xnew, __bf16* __restrict__ h) {
  const int row  = (blockIdx.x * blockDim.x + threadIdx.x) >> 5;
  const int lane = threadIdx.x & 31;
  const int t = row & (TT - 1);  // position within batch (T is power of two)
  const size_t base = (size_t)row * CC;
  float vals[16];
  float s = 0.f;
#pragma unroll
  for (int i = 0; i < 16; ++i) {
    const int c = i * 32 + lane;
    float lv = conv_b[c] + conv_w[c * 3 + 2] * (float)xnorm[base + c];
    if (t >= 1) lv += conv_w[c * 3 + 1] * (float)xnorm[base - CC + c];
    if (t >= 2) lv += conv_w[c * 3 + 0] * (float)xnorm[base - 2 * CC + c];
    const float sl = (float)sig[(size_t)row * (2 * CC) + c];
    const float sg = (float)sig[(size_t)row * (2 * CC) + CC + c];
    const float xv = x[base + c] + sl * lv + sg * (float)gctx[base + c];
    vals[i] = xv;
    s += xv;
  }
  s = wave_sum(s);
  const float mean = s * (1.f / CC);
  float q = 0.f;
#pragma unroll
  for (int i = 0; i < 16; ++i) { float d = vals[i] - mean; q += d * d; }
  q = wave_sum(q);
  const float rstd = rsqrtf(q * (1.f / CC) + 1e-5f);
#pragma unroll
  for (int i = 0; i < 16; ++i) {
    const int c = i * 32 + lane;
    xnew[base + c] = vals[i];
    h[base + c] = (__bf16)((vals[i] - mean) * rstd * g2[c] + b2[c]);
  }
}

// ---------- launch ----------
extern "C" void kernel_launch(void* const* d_in, const int* in_sizes, int n_in,
                              void* d_out, int out_size, void* d_ws,
                              size_t ws_size, hipStream_t stream) {
  (void)in_sizes; (void)n_in; (void)out_size; (void)ws_size;
  const float* x      = (const float*)d_in[0];
  const float* conv_w = (const float*)d_in[1];
  const float* conv_b = (const float*)d_in[2];
  const float* mem    = (const float*)d_in[3];
  const float* gate_w = (const float*)d_in[4];
  const float* gate_b = (const float*)d_in[5];
  const float* w1     = (const float*)d_in[6];
  const float* b1     = (const float*)d_in[7];
  const float* w2     = (const float*)d_in[8];
  const float* b2     = (const float*)d_in[9];
  const float* ln1g   = (const float*)d_in[10];
  const float* ln1b   = (const float*)d_in[11];
  const float* ln2g   = (const float*)d_in[12];
  const float* ln2b   = (const float*)d_in[13];
  float* out = (float*)d_out;

  char* p = (char*)d_ws;
  auto alloc = [&](size_t bytes) -> char* {
    char* r = p;
    p += (bytes + 255) & ~(size_t)255;
    return r;
  };
  __bf16* xnorm = (__bf16*)alloc((size_t)NTOK * CC * 2);
  __bf16* gctx  = (__bf16*)alloc((size_t)NTOK * CC * 2);
  __bf16* sig   = (__bf16*)alloc((size_t)NTOK * 2 * CC * 2);
  float*  xnew  = (float*) alloc((size_t)NTOK * CC * 4);
  __bf16* h     = (__bf16*)alloc((size_t)NTOK * CC * 2);
  __bf16* mid   = (__bf16*)alloc((size_t)NTOK * 4 * CC * 2);
  __bf16* mbB   = (__bf16*)alloc((size_t)512 * 512 * 2);
  __bf16* mbT   = (__bf16*)alloc((size_t)512 * 512 * 2);
  __bf16* gwT   = (__bf16*)alloc((size_t)512 * 1024 * 2);
  __bf16* w1T   = (__bf16*)alloc((size_t)512 * 2048 * 2);
  __bf16* w2T   = (__bf16*)alloc((size_t)2048 * 512 * 2);

  // weight prep (tiny; weights then live in L2)
  cvt_bf16_kernel<<<(512 * 512 + 255) / 256, 256, 0, stream>>>(mem, mbB, 512 * 512);
  transpose_bf16_kernel<<<(512 * 512 + 255) / 256, 256, 0, stream>>>(mem, mbT, 512, 512);
  transpose_bf16_kernel<<<(512 * 1024 + 255) / 256, 256, 0, stream>>>(gate_w, gwT, 512, 1024);
  transpose_bf16_kernel<<<(512 * 2048 + 255) / 256, 256, 0, stream>>>(w1, w1T, 512, 2048);
  transpose_bf16_kernel<<<(2048 * 512 + 255) / 256, 256, 0, stream>>>(w2, w2T, 2048, 512);

  // 1) LN1 -> bf16 x_norm
  ln1_kernel<<<NTOK / 8, 256, 0, stream>>>(x, ln1g, ln1b, xnorm);
  // 2) fused scores/softmax/ctx (scores never touch HBM; A tile async-staged)
  memattn_kernel<<<NTOK / 16, 256, 0, stream>>>(xnorm, mbB, mbT, gctx);
  // 3) gates GEMM + sigmoid
  dim3 gGate(NTOK / 128, 1024 / 64);
  gemm_bf16_kernel<0><<<gGate, 256, 0, stream>>>(xnorm, gwT, gate_b, nullptr,
                                                 sig, nullptr, 512, 1024);
  // 4) conv + gated fusion + residual + LN2
  combine_ln2_kernel<<<NTOK / 8, 256, 0, stream>>>(x, xnorm, gctx, sig, conv_w,
                                                   conv_b, ln2g, ln2b, xnew, h);
  // 5) FFN1 GEMM + gelu
  dim3 gF1(NTOK / 128, 2048 / 64);
  gemm_bf16_kernel<1><<<gF1, 256, 0, stream>>>(h, w1T, b1, nullptr, mid, nullptr,
                                               512, 2048);
  // 6) FFN2 GEMM + bias + residual -> f32 out
  dim3 gF2(NTOK / 128, 512 / 64);
  gemm_bf16_kernel<2><<<gF2, 256, 0, stream>>>(mid, w2T, b2, xnew, nullptr, out,
                                               2048, 512);
}
